// CrossAttention_85718957294226
// MI455X (gfx1250) — matrix operand order repaired and checked by reading, not compiled, not run on hardware
//
#include <hip/hip_runtime.h>

// ---------------------------------------------------------------------------
// CDNA5 (gfx1250) cross-attention, f16 WMMA (v_wmma_f32_16x16x32_f16), wave32
// + async global->LDS staging (ASYNCcnt) + wide global_prefetch_b8
// ---------------------------------------------------------------------------

typedef __attribute__((ext_vector_type(16))) _Float16 v16h;
typedef __attribute__((ext_vector_type(8)))  float    v8f;
typedef __attribute__((ext_vector_type(4)))  float    v4f;

#define D_MODEL 768
#define NSEQ    2048
#define NHEAD   12
#define DKDIM   64
#define BATCH   2
#define MROWS   (BATCH * NSEQ)   // 4096

#if defined(__has_builtin)
#if __has_builtin(__builtin_amdgcn_global_load_async_to_lds_b128)
#define HAVE_ASYNC_LDS 1
#endif
#endif

#ifdef HAVE_ASYNC_LDS
typedef __attribute__((ext_vector_type(4))) int v4i;
typedef __attribute__((address_space(1))) void* as1_vp;
typedef __attribute__((address_space(3))) void* as3_vp;
typedef __attribute__((address_space(1))) v4i*  as1_v4i;
typedef __attribute__((address_space(3))) v4i*  as3_v4i;
#endif

__device__ __forceinline__ void wait_async_le2() {
  asm volatile("s_wait_asynccnt 0x2" ::: "memory");
}

// ---------------------------------------------------------------------------
// WMMA wrapper: D = A(16x32 f16) * B(32x16 f16) + C(16x16 f32)
// ---------------------------------------------------------------------------
__device__ __forceinline__ v8f wmma_f16(v16h a, v16h b, v8f c) {
  return __builtin_amdgcn_wmma_f32_16x16x32_f16(
      /*neg_a=*/false, a, /*neg_b=*/false, b,
      /*c_mod=*/(short)0, c, /*reuse_a=*/false, /*reuse_b=*/false);
}

// A-matrix loader (16x32 f16, row-major source, row stride ld halves).
// ISA layout: lanes 0-15 hold rows M=0..15 with K = 0..7 and 16..23;
//             lanes 16-31 same rows with K = 8..15 and 24..31.
__device__ __forceinline__ v16h load_A16(const _Float16* base, int ld) {
  const int lane = threadIdx.x & 31;
  const int r    = lane & 15;
  const int off  = (lane >> 4) << 3;          // 0 or 8
  const _Float16* p = base + (size_t)r * ld + off;
  union { v16h v; v4f q[2]; } u;
  u.q[0] = *(const v4f*)(p);                  // K = off .. off+7
  u.q[1] = *(const v4f*)(p + 16);             // K = off+16 .. off+23
  return u.v;
}

// B-matrix loader (32x16 f16). Source is "column-contiguous": row n of `base`
// holds all K values of output column n (ld halves apart per column).
// ISA layout: lanes 0-15 hold cols N=0..15 with K = 0..7 and 16..23;
//             lanes 16-31 same cols with K = 8..15 and 24..31.
__device__ __forceinline__ v16h load_B16(const _Float16* base, int ld) {
  const int lane = threadIdx.x & 31;
  const int n    = lane & 15;
  const int off  = (lane >> 4) << 3;          // 0 or 8
  const _Float16* p = base + (size_t)n * ld + off;
  union { v16h v; v4f q[2]; } u;
  u.q[0] = *(const v4f*)(p);                  // K = off .. off+7
  u.q[1] = *(const v4f*)(p + 16);             // K = off+16 .. off+23
  return u.v;
}

// ---------------------------------------------------------------------------
// LayerNorm: one 256-thread block per row of 768, f32 in -> f16 out
// ---------------------------------------------------------------------------
__global__ __launch_bounds__(256)
void ln_kernel(const float* __restrict__ x, const float* __restrict__ w,
               const float* __restrict__ b, _Float16* __restrict__ out) {
  const int row = blockIdx.x;
  const int tid = threadIdx.x;
  const float* xr = x + (size_t)row * D_MODEL;
  __shared__ float red[256];

  float s = 0.f;
  for (int i = tid; i < D_MODEL; i += 256) s += xr[i];
  red[tid] = s; __syncthreads();
  for (int st = 128; st > 0; st >>= 1) {
    if (tid < st) red[tid] += red[tid + st];
    __syncthreads();
  }
  const float mu = red[0] / (float)D_MODEL;
  __syncthreads();

  float v = 0.f;
  for (int i = tid; i < D_MODEL; i += 256) { float d = xr[i] - mu; v += d * d; }
  red[tid] = v; __syncthreads();
  for (int st = 128; st > 0; st >>= 1) {
    if (tid < st) red[tid] += red[tid + st];
    __syncthreads();
  }
  const float rstd = rsqrtf(red[0] / (float)D_MODEL + 1e-5f);

  _Float16* orow = out + (size_t)row * D_MODEL;
  for (int i = tid; i < D_MODEL; i += 256)
    orow[i] = (_Float16)((xr[i] - mu) * rstd * w[i] + b[i]);
}

// ---------------------------------------------------------------------------
// Weight transpose + f16 convert: Wt[n][k] = (f16) W[k][n]   (768x768)
// ---------------------------------------------------------------------------
__global__ __launch_bounds__(256)
void wt_kernel(const float* __restrict__ W, _Float16* __restrict__ Wt) {
  const int idx = blockIdx.x * 256 + threadIdx.x;
  if (idx >= D_MODEL * D_MODEL) return;
  const int n = idx / D_MODEL;
  const int k = idx % D_MODEL;
  Wt[idx] = (_Float16)W[(size_t)k * D_MODEL + n];
}

// ---------------------------------------------------------------------------
// WMMA GEMM: C[M,N] = A[M,K](f16 rowmajor) * Bt[N,K](f16) + bias[N]
// Block = 4 waves sharing one 64-column slab (4 consecutive 16-row tiles).
// The shared 64x32 B tile is staged in LDS, double-buffered, copied with
// global_load_async_to_lds_b128 (ASYNCcnt) so the copy of chunk k+1 overlaps
// the WMMAs of chunk k. Output modes:
//   0: f16 row-major [M,N]
//   1: f16 heads      [B,H,NSEQ,DK]   (for Q,K)
//   2: f16 heads^T    [B,H,DK,NSEQ]   (for V)
//   3: f32 row-major [M,N]            (final outputs)
// ---------------------------------------------------------------------------
__global__ __launch_bounds__(128)
void gemm_kernel(const _Float16* __restrict__ A, const _Float16* __restrict__ Bt,
                 const float* __restrict__ bias, void* __restrict__ outp,
                 int M, int N, int K, int mode) {
  const int tid  = threadIdx.x;
  const int lane = tid & 31;
  const int half = lane >> 4;
  const int l16  = lane & 15;
  const int w    = blockIdx.x * 4 + (tid >> 5);
  const int mtiles = M >> 4;                 // 256 (divisible by 4: no nt straddle)
  const int mt = w % mtiles;
  const int nt = w / mtiles;
  const int colbase = nt * 64;

  // double-buffered B slab: 64 output columns x 32 k-values, f16
  __shared__ __align__(16) _Float16 bufB[2][64 * 32];

  // cooperative copy of one 4KB slab: 128 threads x 2 x 16B
  auto issue_copy = [&](int kc, int buf) {
#pragma unroll
    for (int c = 0; c < 2; ++c) {
      const int byteoff = tid * 16 + c * 2048;
      const int row     = byteoff >> 6;        // 64 B per output column
      const int halfoff = (byteoff & 63) >> 1;
      const _Float16* src = Bt + (size_t)(colbase + row) * K + kc + halfoff;
      _Float16*       dst = &bufB[buf][0] + (byteoff >> 1);
#ifdef HAVE_ASYNC_LDS
      __builtin_amdgcn_global_load_async_to_lds_b128(
          (as1_v4i)(as1_vp)(void*)src, (as3_v4i)(as3_vp)(void*)dst, 0, 0);
#else
      *(v4f*)dst = *(const v4f*)src;           // sync fallback: b128 + ds_store
#endif
    }
  };

  v8f c[4];
#pragma unroll
  for (int t = 0; t < 4; ++t) {
    const float bv = bias[colbase + t * 16 + l16];
#pragma unroll
    for (int j = 0; j < 8; ++j) c[t][j] = bv;
  }

  const _Float16* arow = A + (size_t)(mt * 16) * K;
  const int nch = K / 32;                     // 24

  issue_copy(0, 0);
  for (int i = 0; i < nch; ++i) {
    __syncthreads();                          // buf[(i+1)&1] free for rewrite
    // prefetch next chunk (last iteration harmlessly re-reads chunk 0 into
    // the dead buffer; drained by S_ENDPGM's implicit wait-idle)
    issue_copy((i + 1 < nch) ? (i + 1) * 32 : 0, (i + 1) & 1);
#ifdef HAVE_ASYNC_LDS
    wait_async_le2();                         // chunk i landed, i+1 in flight
#endif
    __syncthreads();                          // everyone's chunk-i data visible

    v16h a = load_A16(arow + i * 32, K);
    const _Float16* bb = &bufB[i & 1][0];
    v16h b[4];
#pragma unroll
    for (int t = 0; t < 4; ++t) b[t] = load_B16(bb + t * 16 * 32, 32);
#pragma unroll
    for (int t = 0; t < 4; ++t) c[t] = wmma_f16(a, b[t], c[t]);
  }

#pragma unroll
  for (int t = 0; t < 4; ++t) {
#pragma unroll
    for (int j = 0; j < 8; ++j) {
      const int row = mt * 16 + j + half * 8;
      const int col = colbase + t * 16 + l16;
      const float val = c[t][j];
      if (mode == 3) {
        ((float*)outp)[(size_t)row * N + col] = val;
      } else if (mode == 0) {
        ((_Float16*)outp)[(size_t)row * N + col] = (_Float16)val;
      } else {
        const int bb = row >> 11;          // row / NSEQ
        const int n  = row & (NSEQ - 1);
        const int hh = col >> 6;           // col / DK
        const int d  = col & (DKDIM - 1);
        if (mode == 1)
          ((_Float16*)outp)[(((size_t)(bb * NHEAD + hh)) * NSEQ + n) * DKDIM + d] =
              (_Float16)val;
        else  // mode 2: transposed per head
          ((_Float16*)outp)[(((size_t)(bb * NHEAD + hh)) * DKDIM + d) * NSEQ + n] =
              (_Float16)val;
      }
    }
  }
}

// ---------------------------------------------------------------------------
// Flash-style attention. One wave (32 threads) per (b, h, 16-query-row block).
//   Q,Kk : [B,H,NSEQ,DK] f16     Vt : [B,H,DK,NSEQ] f16
//   Out  : [B,NSEQ,D_MODEL] f16  (heads re-interleaved)
// Per-lane wide prefetch (global_prefetch_b8) pulls K/V two blocks ahead.
// ---------------------------------------------------------------------------
__global__ __launch_bounds__(32)
void attn_kernel(const _Float16* __restrict__ Q, const _Float16* __restrict__ Kk,
                 const _Float16* __restrict__ Vt, _Float16* __restrict__ Out) {
  const int lane = threadIdx.x & 31;
  const int half = lane >> 4;
  const int l16  = lane & 15;

  const int mtiles = NSEQ / 16;
  const int mt = blockIdx.x % mtiles;
  const int bh = blockIdx.x / mtiles;
  const int h  = bh % NHEAD;
  const int b  = bh / NHEAD;

  const _Float16* qbase = Q  + (((size_t)(b * NHEAD + h)) * NSEQ + mt * 16) * DKDIM;
  const _Float16* kbase = Kk + ((size_t)(b * NHEAD + h)) * NSEQ * DKDIM;
  const _Float16* vbase = Vt + ((size_t)(b * NHEAD + h)) * DKDIM * NSEQ;

  // Q row-block, cached for the whole loop (two K-chunks of 32)
  const v16h a0 = load_A16(qbase, DKDIM);
  const v16h a1 = load_A16(qbase + 32, DKDIM);

  v8f o[4];
  float m[8], l[8];
#pragma unroll
  for (int t = 0; t < 4; ++t)
#pragma unroll
    for (int j = 0; j < 8; ++j) o[t][j] = 0.f;
#pragma unroll
  for (int j = 0; j < 8; ++j) { m[j] = -1e30f; l[j] = 0.f; }

  __shared__ __align__(16) _Float16 pbuf[16 * 48];   // 16x32 P tile, stride 48

  const float scale = 0.125f;  // 1/sqrt(64)

  for (int kb = 0; kb < NSEQ; kb += 32) {
    // ---- wide prefetch two key-blocks ahead (one cacheline per lane) ----
    if (kb + 64 < NSEQ) {
      __builtin_prefetch(kbase + (size_t)(kb + 64) * DKDIM + (size_t)lane * 64, 0, 3);
      __builtin_prefetch(vbase + (size_t)lane * NSEQ + (kb + 64), 0, 3);
      __builtin_prefetch(vbase + (size_t)(32 + lane) * NSEQ + (kb + 64), 0, 3);
    }

    // ---- S = Q K^T for 32 keys (two 16-col tiles, dk=64 = 2 chunks) ----
    v8f s0 = {}, s1 = {};
    {
      v16h b00 = load_B16(kbase + (size_t)kb * DKDIM,             DKDIM);
      v16h b01 = load_B16(kbase + (size_t)kb * DKDIM + 32,        DKDIM);
      v16h b10 = load_B16(kbase + (size_t)(kb + 16) * DKDIM,      DKDIM);
      v16h b11 = load_B16(kbase + (size_t)(kb + 16) * DKDIM + 32, DKDIM);
      s0 = wmma_f16(a0, b00, s0);
      s0 = wmma_f16(a1, b01, s0);
      s1 = wmma_f16(a0, b10, s1);
      s1 = wmma_f16(a1, b11, s1);
    }

    // ---- online softmax (rows live across 16-lane half-waves) ----
    float alpha[8];
#pragma unroll
    for (int j = 0; j < 8; ++j) {
      s0[j] *= scale;
      s1[j] *= scale;
      float v = fmaxf(s0[j], s1[j]);
      v = fmaxf(v, __shfl_xor(v, 8, 32));
      v = fmaxf(v, __shfl_xor(v, 4, 32));
      v = fmaxf(v, __shfl_xor(v, 2, 32));
      v = fmaxf(v, __shfl_xor(v, 1, 32));
      const float mn = fmaxf(m[j], v);
      alpha[j] = __expf(m[j] - mn);
      m[j] = mn;
      s0[j] = __expf(s0[j] - mn);
      s1[j] = __expf(s1[j] - mn);
      float rs = s0[j] + s1[j];
      rs += __shfl_xor(rs, 8, 32);
      rs += __shfl_xor(rs, 4, 32);
      rs += __shfl_xor(rs, 2, 32);
      rs += __shfl_xor(rs, 1, 32);
      l[j] = l[j] * alpha[j] + rs;
    }
#pragma unroll
    for (int t = 0; t < 4; ++t)
#pragma unroll
      for (int j = 0; j < 8; ++j) o[t][j] *= alpha[j];

    // ---- transpose P: C-layout -> A-layout through LDS ----
    __syncthreads();
#pragma unroll
    for (int j = 0; j < 8; ++j) {
      pbuf[(j + half * 8) * 48 + l16]      = (_Float16)s0[j];
      pbuf[(j + half * 8) * 48 + 16 + l16] = (_Float16)s1[j];
    }
    __syncthreads();
    v16h pA = load_A16(pbuf, 48);

    // ---- O += P * V (four 16-col dim tiles; loads hoisted ahead of WMMAs) ----
    v16h bv[4];
#pragma unroll
    for (int t = 0; t < 4; ++t)
      bv[t] = load_B16(vbase + (size_t)(t * 16) * NSEQ + kb, NSEQ);
#pragma unroll
    for (int t = 0; t < 4; ++t) o[t] = wmma_f16(pA, bv[t], o[t]);
  }

  // ---- finalize + store (re-interleave heads into [B,N,D]) ----
#pragma unroll
  for (int j = 0; j < 8; ++j) {
    const float inv = 1.0f / l[j];
#pragma unroll
    for (int t = 0; t < 4; ++t) o[t][j] *= inv;
  }
#pragma unroll
  for (int t = 0; t < 4; ++t)
#pragma unroll
    for (int j = 0; j < 8; ++j) {
      const int row = mt * 16 + j + half * 8;
      const int col = h * DKDIM + t * 16 + l16;
      Out[((size_t)b * NSEQ + row) * D_MODEL + col] = (_Float16)o[t][j];
    }
}

// ---------------------------------------------------------------------------
// Host-side orchestration
// ---------------------------------------------------------------------------
extern "C" void kernel_launch(void* const* d_in, const int* in_sizes, int n_in,
                              void* d_out, int out_size, void* d_ws, size_t ws_size,
                              hipStream_t stream) {
  (void)in_sizes; (void)n_in; (void)out_size; (void)ws_size;

  const float* rgb   = (const float*)d_in[0];
  const float* ir    = (const float*)d_in[1];
  const float* ln0w  = (const float*)d_in[2];
  const float* ln0b  = (const float*)d_in[3];
  const float* ln1w  = (const float*)d_in[4];
  const float* ln1b  = (const float*)d_in[5];

  // workspace layout (all f16 regions)
  char* ws = (char*)d_ws;
  const size_t szAct = (size_t)MROWS * D_MODEL * sizeof(_Float16);       // 6.29 MB
  const size_t szW   = (size_t)D_MODEL * D_MODEL * sizeof(_Float16);     // 1.18 MB
  _Float16* rgbn = (_Float16*)(ws + 0 * szAct);
  _Float16* irn  = (_Float16*)(ws + 1 * szAct);
  _Float16* qv   = (_Float16*)(ws + 2 * szAct);
  _Float16* kv   = (_Float16*)(ws + 3 * szAct);
  _Float16* vv   = (_Float16*)(ws + 4 * szAct);
  _Float16* qi   = (_Float16*)(ws + 5 * szAct);
  _Float16* ki   = (_Float16*)(ws + 6 * szAct);
  _Float16* vi   = (_Float16*)(ws + 7 * szAct);
  _Float16* aov  = (_Float16*)(ws + 8 * szAct);
  _Float16* aoi  = (_Float16*)(ws + 9 * szAct);
  _Float16* Wt[8];
  for (int i = 0; i < 8; ++i) Wt[i] = (_Float16*)(ws + 10 * szAct + i * szW);

  // 1) LayerNorm -> f16 activations
  ln_kernel<<<MROWS, 256, 0, stream>>>(rgb, ln0w, ln0b, rgbn);
  ln_kernel<<<MROWS, 256, 0, stream>>>(ir,  ln1w, ln1b, irn);

  // 2) Transpose + convert weights to f16
  const int wtBlocks = (D_MODEL * D_MODEL + 255) / 256;
  const int widx[8] = {6, 8, 10, 12, 14, 16, 18, 20};  // Wq_vis,Wk_vis,Wq_ir,Wk_ir,Wv_vis,Wv_ir,Wo_vis,Wo_ir
  for (int i = 0; i < 8; ++i)
    wt_kernel<<<wtBlocks, 256, 0, stream>>>((const float*)d_in[widx[i]], Wt[i]);

  // 3) Projections (16x64 tile per wave; 3072 tiles -> 768 blocks of 4 waves)
  const int gemmBlocks = (MROWS / 16) * (D_MODEL / 64) / 4;
  gemm_kernel<<<gemmBlocks, 128, 0, stream>>>(rgbn, Wt[0], (const float*)d_in[7],  qv, MROWS, D_MODEL, D_MODEL, 1);
  gemm_kernel<<<gemmBlocks, 128, 0, stream>>>(rgbn, Wt[1], (const float*)d_in[9],  kv, MROWS, D_MODEL, D_MODEL, 1);
  gemm_kernel<<<gemmBlocks, 128, 0, stream>>>(rgbn, Wt[4], (const float*)d_in[15], vv, MROWS, D_MODEL, D_MODEL, 2);
  gemm_kernel<<<gemmBlocks, 128, 0, stream>>>(irn,  Wt[2], (const float*)d_in[11], qi, MROWS, D_MODEL, D_MODEL, 1);
  gemm_kernel<<<gemmBlocks, 128, 0, stream>>>(irn,  Wt[3], (const float*)d_in[13], ki, MROWS, D_MODEL, D_MODEL, 1);
  gemm_kernel<<<gemmBlocks, 128, 0, stream>>>(irn,  Wt[5], (const float*)d_in[17], vi, MROWS, D_MODEL, D_MODEL, 2);

  // 4) Cross attention: out_vis uses (q_ir, k_vis, v_vis); out_ir uses (q_vis, k_ir, v_ir)
  const int attnBlocks = BATCH * NHEAD * (NSEQ / 16);   // 3072 single-wave WGs
  attn_kernel<<<attnBlocks, 32, 0, stream>>>(qi, kv, vv, aov);
  attn_kernel<<<attnBlocks, 32, 0, stream>>>(qv, ki, vi, aoi);

  // 5) Output projections -> f32 d_out (out_vis then out_ir, concatenated)
  float* out_vis = (float*)d_out;
  float* out_ir  = out_vis + (size_t)MROWS * D_MODEL;
  gemm_kernel<<<gemmBlocks, 128, 0, stream>>>(aov, Wt[6], (const float*)d_in[19], out_vis, MROWS, D_MODEL, D_MODEL, 3);
  gemm_kernel<<<gemmBlocks, 128, 0, stream>>>(aoi, Wt[7], (const float*)d_in[21], out_ir,  MROWS, D_MODEL, D_MODEL, 3);
}
